// ProbabilityAttention_20882130993615
// MI455X (gfx1250) — compile-verified, matrix operand order checked by reading, not run on hardware
//
#include <hip/hip_runtime.h>
#include <hip/hip_bf16.h>

// ---------------- problem constants (from reference) ----------------
#define BB   4
#define LL   2048
#define EE   512
#define HH   8
#define DD   64
#define RR   40          // rand_num
#define TOP  35          // top_num
#define TOPP 48          // padded to 3 WMMA row tiles
#define BH   (BB*HH)     // 32
#define BL   (BB*LL)     // 8192
#define SCALE 0.125f     // 1/sqrt(64)
#define NEG_INF (-3.0e38f)

typedef __attribute__((ext_vector_type(16))) __bf16 v16bf;
typedef __attribute__((ext_vector_type(8)))  __bf16 v8bf;   // one 16-byte fragment half
typedef __attribute__((ext_vector_type(8)))  float  v8f;

union Frag { v16bf f; v8bf h[2]; };

// Optional gfx1250 async-copy-to-LDS path (guarded: falls back to direct b128 loads)
#if defined(__has_builtin)
#  if __has_builtin(__builtin_amdgcn_global_load_async_to_lds_b128) && \
      __has_builtin(__builtin_amdgcn_s_wait_asynccnt)
#    define PA_HAVE_ASYNC 1
#  endif
#endif

#ifdef PA_HAVE_ASYNC
// builtin takes typed b128 pointers: int vector_size(16) in AS(1)/AS(3)
typedef int pa_v4i __attribute__((vector_size(16)));
typedef __attribute__((address_space(1))) pa_v4i* pa_gptr128;
typedef __attribute__((address_space(3))) pa_v4i* pa_lptr128;
#endif

__device__ __forceinline__ v8f wmma_bf16(v16bf a, v16bf b, v8f c) {
    return __builtin_amdgcn_wmma_f32_16x16x32_bf16(
        false, a, false, b, (short)0, c, false, false);
}

// A/B^T fragment of a 16x32 bf16 tile: lane half `half` owns K runs
// [half*8, half*8+8) and [16+half*8, 16+half*8+8) -> two aligned 16B loads.
__device__ __forceinline__ v16bf load_frag(const __bf16* __restrict__ row, int half) {
    Frag u;
    u.h[0] = *(const v8bf*)(row + half * 8);
    u.h[1] = *(const v8bf*)(row + 16 + half * 8);
    return u.f;
}
__device__ __forceinline__ v16bf frag_from_lds(const __bf16* p) {
    Frag u;
    u.h[0] = *(const v8bf*)(p);
    u.h[1] = *(const v8bf*)(p + 8);
    return u.f;
}

// ---------------------------------------------------------------
// K0a: weights fp32 [K,N] -> bf16 transposed [N,K]   grid=(512,4) block=256
// ---------------------------------------------------------------
__global__ __launch_bounds__(256)
void pa_conv_weights(const float* __restrict__ Wq, const float* __restrict__ Wk,
                     const float* __restrict__ Wv, const float* __restrict__ Wo,
                     __bf16* __restrict__ Wt) {
    const int n = blockIdx.x, w = blockIdx.y, t = threadIdx.x;
    const float* W = (w == 0) ? Wq : (w == 1) ? Wk : (w == 2) ? Wv : Wo;
    __bf16* dst = Wt + ((size_t)w * EE + n) * EE;
    for (int k = t; k < EE; k += 256) dst[k] = (__bf16)W[(size_t)k * EE + n];
}

// ---------------------------------------------------------------
// K0b: activations fp32 -> bf16 row-major   grid=(8192,3) block=256
// ---------------------------------------------------------------
__global__ __launch_bounds__(256)
void pa_conv_x(const float* __restrict__ Xq, const float* __restrict__ Xk,
               const float* __restrict__ Xv, __bf16* __restrict__ Xbf) {
    const int row = blockIdx.x, z = blockIdx.y, t = threadIdx.x;
    const float* X = (z == 0) ? Xq : (z == 1) ? Xk : Xv;
    __bf16* dst = Xbf + ((size_t)z * BL + row) * EE;
    for (int e = t; e < EE; e += 256) dst[e] = (__bf16)X[(size_t)row * EE + e];
}

// ---------------------------------------------------------------
// K1: fused Q/K/V projection GEMM, 32x32 tile per wave (2x2 WMMA tiles),
//     LDS double-buffered via GLOBAL_LOAD_ASYNC_TO_LDS when available.
// grid = (16, 256, 3), block = 32
// ---------------------------------------------------------------
__global__ __launch_bounds__(32)
void pa_qkv_gemm(const __bf16* __restrict__ Xbf, const __bf16* __restrict__ Wt,
                 const float* __restrict__ bq, const float* __restrict__ bk,
                 const float* __restrict__ bv,
                 float* __restrict__ qws, float* __restrict__ kws,
                 float* __restrict__ vws,
                 __bf16* __restrict__ kbf, __bf16* __restrict__ vtb) {
    const int z = blockIdx.z;
    const __bf16* X = Xbf + (size_t)z * BL * EE;
    const __bf16* W = Wt + (size_t)z * EE * EE;
    const float* bias = (z == 0) ? bq : (z == 1) ? bk : bv;
    float* Y = (z == 0) ? qws : (z == 1) ? kws : vws;

    const int tn = blockIdx.x, tm = blockIdx.y;
    const int lane = threadIdx.x;
    const int m = lane & 15, half = lane >> 4;

    const __bf16* arow0 = X + (size_t)(tm * 32 + m) * EE;
    const __bf16* arow1 = arow0 + (size_t)16 * EE;
    const __bf16* brow0 = W + (size_t)(tn * 32 + m) * EE;
    const __bf16* brow1 = brow0 + (size_t)16 * EE;

    v8f acc00 = {}, acc01 = {}, acc10 = {}, acc11 = {};

#ifdef PA_HAVE_ASYNC
    // 2 buffers x 32 lanes x 8 slots x 16B = 8 KB
    __shared__ __align__(16) __bf16 stg[2][32][64];
    __bf16* lds0 = &stg[0][lane][0];
    __bf16* lds1 = &stg[1][lane][0];

    auto stage = [&](__bf16* L, int kt) {
        const __bf16* g[8] = {
            arow0 + kt + half * 8, arow0 + kt + 16 + half * 8,
            arow1 + kt + half * 8, arow1 + kt + 16 + half * 8,
            brow0 + kt + half * 8, brow0 + kt + 16 + half * 8,
            brow1 + kt + half * 8, brow1 + kt + 16 + half * 8 };
#pragma unroll
        for (int s = 0; s < 8; ++s)
            __builtin_amdgcn_global_load_async_to_lds_b128(
                (pa_gptr128)g[s],
                (pa_lptr128)(L + s * 8),
                0, 0);
    };

    stage(lds0, 0);
    int buf = 0;
    for (int kt = 0; kt < EE; kt += 32) {
        __bf16* cur = buf ? lds1 : lds0;
        __bf16* nxt = buf ? lds0 : lds1;
        if (kt + 32 < EE) {
            stage(nxt, kt + 32);
            __builtin_amdgcn_s_wait_asynccnt(8);   // current buffer complete
        } else {
            __builtin_amdgcn_s_wait_asynccnt(0);
        }
        asm volatile("" ::: "memory");
        v16bf a0 = frag_from_lds(cur +  0);
        v16bf a1 = frag_from_lds(cur + 16);
        v16bf b0 = frag_from_lds(cur + 32);
        v16bf b1 = frag_from_lds(cur + 48);
        acc00 = wmma_bf16(a0, b0, acc00);
        acc01 = wmma_bf16(a0, b1, acc01);
        acc10 = wmma_bf16(a1, b0, acc10);
        acc11 = wmma_bf16(a1, b1, acc11);
        buf ^= 1;
    }
#else
    for (int kt = 0; kt < EE; kt += 32) {
        __builtin_prefetch(arow0 + kt + 32, 0, 1);
        v16bf a0 = load_frag(arow0 + kt, half);
        v16bf a1 = load_frag(arow1 + kt, half);
        v16bf b0 = load_frag(brow0 + kt, half);
        v16bf b1 = load_frag(brow1 + kt, half);
        acc00 = wmma_bf16(a0, b0, acc00);
        acc01 = wmma_bf16(a0, b1, acc01);
        acc10 = wmma_bf16(a1, b0, acc10);
        acc11 = wmma_bf16(a1, b1, acc11);
    }
#endif

    auto store16 = [&](const v8f& acc, int gm0, int gn0) {
#pragma unroll
        for (int r = 0; r < 8; ++r) {
            const int gm = gm0 + half * 8 + r;        // b*L + l
            const int gn = gn0 + m;                   // h*64 + d
            const int bb = gm >> 11, l = gm & (LL - 1);
            const int h = gn >> 6, d = gn & 63;
            const size_t idx = ((((size_t)bb * HH + h) * LL) + l) * DD + d;
            const float val = acc[r] + bias[gn];
            Y[idx] = val;
            if (z == 1) kbf[idx] = (__bf16)val;
            if (z == 2) vtb[(((size_t)(bb * HH + h) * DD) + d) * LL + l] = (__bf16)val;
        }
    };
    store16(acc00, tm * 32,      tn * 32);
    store16(acc01, tm * 32,      tn * 32 + 16);
    store16(acc10, tm * 32 + 16, tn * 32);
    store16(acc11, tm * 32 + 16, tn * 32 + 16);
}

// ---------------------------------------------------------------
// K2: sampled scores -> discrete = max_r(q.kr) - sum_r(q.kr)/L
// grid = (L, H, B), block = 32
// ---------------------------------------------------------------
__global__ __launch_bounds__(32)
void pa_sample_scores(const float* __restrict__ qws, const float* __restrict__ kws,
                      const int* __restrict__ rand_index, float* __restrict__ disc) {
    const int qpos = blockIdx.x;
    const int bh = blockIdx.z * HH + blockIdx.y;
    const int lane = threadIdx.x;
    const float* qrow = qws + ((size_t)bh * LL + qpos) * DD;   // uniform address

    float best = NEG_INF, ssum = 0.0f;
    for (int r = lane; r < RR; r += 32) {
        const int kidx = rand_index[qpos * RR + r];
        const float* krow = kws + ((size_t)bh * LL + kidx) * DD;
        float dot = 0.0f;
#pragma unroll
        for (int d = 0; d < DD; ++d) dot += qrow[d] * krow[d];
        best = fmaxf(best, dot);
        ssum += dot;
    }
#pragma unroll
    for (int off = 16; off > 0; off >>= 1) {
        best = fmaxf(best, __shfl_xor(best, off, 32));
        ssum += __shfl_xor(ssum, off, 32);
    }
    if (lane == 0)
        disc[(size_t)bh * LL + qpos] = best - ssum * (1.0f / (float)LL);
}

// ---------------------------------------------------------------
// K3: per-(b,h) top-35 via iterative argmax.  grid = 32, block = 256
// ---------------------------------------------------------------
__global__ __launch_bounds__(256)
void pa_topk(const float* __restrict__ disc, int* __restrict__ topidx) {
    __shared__ float vals[LL];
    __shared__ float rv[256];
    __shared__ int   ri[256];
    const int bh = blockIdx.x, t = threadIdx.x;
    for (int i = t; i < LL; i += 256) vals[i] = disc[(size_t)bh * LL + i];
    __syncthreads();
    for (int sel = 0; sel < TOP; ++sel) {
        float bv = NEG_INF; int bi = 0;
        for (int i = t; i < LL; i += 256) {
            const float v = vals[i];
            if (v > bv) { bv = v; bi = i; }
        }
        rv[t] = bv; ri[t] = bi;
        __syncthreads();
        for (int s = 128; s > 0; s >>= 1) {
            if (t < s) {
                if (rv[t + s] > rv[t] || (rv[t + s] == rv[t] && ri[t + s] < ri[t])) {
                    rv[t] = rv[t + s]; ri[t] = ri[t + s];
                }
            }
            __syncthreads();
        }
        if (t == 0) { topidx[bh * TOPP + sel] = ri[0]; vals[ri[0]] = NEG_INF; }
        __syncthreads();
    }
    if (t >= TOP && t < TOPP) topidx[bh * TOPP + t] = -1;   // padded rows
}

// ---------------------------------------------------------------
// K4: gather selected query rows into bf16 (zeros for padding)
// grid = (48, 32), block = 64
// ---------------------------------------------------------------
__global__ __launch_bounds__(64)
void pa_gather_q(const float* __restrict__ qws, const int* __restrict__ topidx,
                 __bf16* __restrict__ qimpbf) {
    const int tpos = blockIdx.x, bh = blockIdx.y, d = threadIdx.x;
    const int idx = topidx[bh * TOPP + tpos];
    qimpbf[((size_t)bh * TOPP + tpos) * DD + d] =
        (idx >= 0) ? (__bf16)qws[((size_t)bh * LL + idx) * DD + d] : (__bf16)0.0f;
}

// ---------------------------------------------------------------
// K5: S = scale * Qimp @ K^T, 16x64 per wave.  grid = (32, 3, 32), block = 32
// ---------------------------------------------------------------
__global__ __launch_bounds__(32)
void pa_scores_wmma(const __bf16* __restrict__ qimpbf, const __bf16* __restrict__ kbf,
                    float* __restrict__ S) {
    const int tn = blockIdx.x, tm = blockIdx.y, bh = blockIdx.z;
    const int lane = threadIdx.x;
    const int m = lane & 15, half = lane >> 4;
    const __bf16* arow = qimpbf + ((size_t)bh * TOPP + tm * 16 + m) * DD;
    const __bf16* brow[4];
#pragma unroll
    for (int j = 0; j < 4; ++j)
        brow[j] = kbf + ((size_t)bh * LL + tn * 64 + j * 16 + m) * DD;

    v8f acc[4] = {};
#pragma unroll
    for (int kt = 0; kt < DD; kt += 32) {
        const v16bf a = load_frag(arow + kt, half);
#pragma unroll
        for (int j = 0; j < 4; ++j)
            acc[j] = wmma_bf16(a, load_frag(brow[j] + kt, half), acc[j]);
    }
#pragma unroll
    for (int j = 0; j < 4; ++j)
#pragma unroll
        for (int r = 0; r < 8; ++r) {
            const int mm = tm * 16 + half * 8 + r;
            const int nn = tn * 64 + j * 16 + m;
            S[((size_t)bh * TOPP + mm) * LL + nn] = acc[j][r] * SCALE;
        }
}

// ---------------------------------------------------------------
// K6: row softmax over 2048 (fp32 in place + normalized bf16 copy)
// grid = 1536, block = 256
// ---------------------------------------------------------------
__global__ __launch_bounds__(256)
void pa_softmax_rows(float* __restrict__ S, __bf16* __restrict__ Pbf) {
    __shared__ float red[256];
    const int t = threadIdx.x;
    float* p = S + (size_t)blockIdx.x * LL;
    __bf16* pb = Pbf + (size_t)blockIdx.x * LL;

    float mx = NEG_INF;
    for (int i = t; i < LL; i += 256) mx = fmaxf(mx, p[i]);
    red[t] = mx; __syncthreads();
    for (int s = 128; s > 0; s >>= 1) { if (t < s) red[t] = fmaxf(red[t], red[t + s]); __syncthreads(); }
    mx = red[0]; __syncthreads();

    float sum = 0.0f;
    for (int i = t; i < LL; i += 256) { const float e = __expf(p[i] - mx); p[i] = e; sum += e; }
    red[t] = sum; __syncthreads();
    for (int s = 128; s > 0; s >>= 1) { if (t < s) red[t] += red[t + s]; __syncthreads(); }
    const float inv = 1.0f / red[0];
    for (int i = t; i < LL; i += 256) {
        const float v = p[i] * inv;
        p[i] = v;
        pb[i] = (__bf16)v;
    }
}

// ---------------------------------------------------------------
// K7: upd = P @ V, 16x64 per wave (V accessed via transposed bf16 vtb)
// grid = (3, 32), block = 32
// ---------------------------------------------------------------
__global__ __launch_bounds__(32)
void pa_upd_wmma(const __bf16* __restrict__ Pbf, const __bf16* __restrict__ vtb,
                 float* __restrict__ upd) {
    const int tm = blockIdx.x, bh = blockIdx.y;
    const int lane = threadIdx.x;
    const int m = lane & 15, half = lane >> 4;
    const __bf16* arow = Pbf + ((size_t)bh * TOPP + tm * 16 + m) * LL;
    const __bf16* brow[4];
#pragma unroll
    for (int j = 0; j < 4; ++j)
        brow[j] = vtb + ((size_t)bh * DD + j * 16 + m) * LL;

    v8f acc[4] = {};
    for (int kt = 0; kt < LL; kt += 32) {
        __builtin_prefetch(arow + kt + 32, 0, 1);
        const v16bf a = load_frag(arow + kt, half);
#pragma unroll
        for (int j = 0; j < 4; ++j)
            acc[j] = wmma_bf16(a, load_frag(brow[j] + kt, half), acc[j]);
    }
#pragma unroll
    for (int j = 0; j < 4; ++j)
#pragma unroll
        for (int r = 0; r < 8; ++r) {
            const int mm = tm * 16 + half * 8 + r;
            const int nn = j * 16 + m;
            upd[((size_t)bh * TOPP + mm) * DD + nn] = acc[j][r];
        }
}

// ---------------------------------------------------------------
// K8: vc = cumsum(v, axis=L).  grid = 32, block = 64
// ---------------------------------------------------------------
__global__ __launch_bounds__(64)
void pa_cumsum(const float* __restrict__ vws, float* __restrict__ vc) {
    const int bh = blockIdx.x, d = threadIdx.x;
    float s = 0.0f;
    for (int l = 0; l < LL; ++l) {
        s += vws[((size_t)bh * LL + l) * DD + d];
        vc[((size_t)bh * LL + l) * DD + d] = s;
    }
}

// ---------------------------------------------------------------
// K9: scatter attention updates into vc.  grid = (35, 32), block = 64
// ---------------------------------------------------------------
__global__ __launch_bounds__(64)
void pa_scatter(const int* __restrict__ topidx, const float* __restrict__ upd,
                float* __restrict__ vc) {
    const int t = blockIdx.x, bh = blockIdx.y, d = threadIdx.x;
    const int row = topidx[bh * TOPP + t];
    if (row >= 0)
        vc[((size_t)bh * LL + row) * DD + d] = upd[((size_t)bh * TOPP + t) * DD + d];
}

// ---------------------------------------------------------------
// K9b: pack vc [B,H,L,D] fp32 -> [B*L, E] bf16 rows for the output GEMM
// grid = (2048, 4), block = 256
// ---------------------------------------------------------------
__global__ __launch_bounds__(256)
void pa_pack_vc(const float* __restrict__ vc, __bf16* __restrict__ vcbf) {
    const int l = blockIdx.x, b = blockIdx.y, t = threadIdx.x;
    for (int e = t; e < EE; e += 256) {
        const int h = e >> 6, d = e & 63;
        vcbf[((size_t)b * LL + l) * EE + e] =
            (__bf16)vc[(((size_t)(b * HH + h)) * LL + l) * DD + d];
    }
}

// ---------------------------------------------------------------
// K10: out = vcbf @ Wo + bo, 32x32 per wave.  grid = (16, 256), block = 32
// ---------------------------------------------------------------
__global__ __launch_bounds__(32)
void pa_out_gemm(const __bf16* __restrict__ vcbf, const __bf16* __restrict__ Wot,
                 const float* __restrict__ bo, float* __restrict__ out) {
    const int tn = blockIdx.x, tm = blockIdx.y;
    const int lane = threadIdx.x;
    const int m = lane & 15, half = lane >> 4;

    const __bf16* arow0 = vcbf + (size_t)(tm * 32 + m) * EE;
    const __bf16* arow1 = arow0 + (size_t)16 * EE;
    const __bf16* brow0 = Wot + (size_t)(tn * 32 + m) * EE;
    const __bf16* brow1 = brow0 + (size_t)16 * EE;

    v8f acc00 = {}, acc01 = {}, acc10 = {}, acc11 = {};
    for (int kt = 0; kt < EE; kt += 32) {
        __builtin_prefetch(arow0 + kt + 32, 0, 1);
        v16bf a0 = load_frag(arow0 + kt, half);
        v16bf a1 = load_frag(arow1 + kt, half);
        v16bf b0 = load_frag(brow0 + kt, half);
        v16bf b1 = load_frag(brow1 + kt, half);
        acc00 = wmma_bf16(a0, b0, acc00);
        acc01 = wmma_bf16(a0, b1, acc01);
        acc10 = wmma_bf16(a1, b0, acc10);
        acc11 = wmma_bf16(a1, b1, acc11);
    }

    auto store16 = [&](const v8f& acc, int gm0, int gn0) {
#pragma unroll
        for (int r = 0; r < 8; ++r) {
            const int gm = gm0 + half * 8 + r;
            const int gn = gn0 + m;
            out[(size_t)gm * EE + gn] = acc[r] + bo[gn];
        }
    };
    store16(acc00, tm * 32,      tn * 32);
    store16(acc01, tm * 32,      tn * 32 + 16);
    store16(acc10, tm * 32 + 16, tn * 32);
    store16(acc11, tm * 32 + 16, tn * 32 + 16);
}

// ---------------------------------------------------------------
extern "C" void kernel_launch(void* const* d_in, const int* in_sizes, int n_in,
                              void* d_out, int out_size, void* d_ws, size_t ws_size,
                              hipStream_t stream) {
    const float* queries = (const float*)d_in[0];
    const float* keys    = (const float*)d_in[1];
    const float* values  = (const float*)d_in[2];
    const float* Wq = (const float*)d_in[3];
    const float* bq = (const float*)d_in[4];
    const float* Wk = (const float*)d_in[5];
    const float* bk = (const float*)d_in[6];
    const float* Wv = (const float*)d_in[7];
    const float* bv = (const float*)d_in[8];
    const float* Wo = (const float*)d_in[9];
    const float* bo = (const float*)d_in[10];
    const int* rand_index = (const int*)d_in[11];
    float* out = (float*)d_out;

    char* ws = (char*)d_ws;
    size_t off = 0;
    auto carve = [&](size_t bytes) -> void* {
        void* p = ws + off;
        off += (bytes + 255) & ~(size_t)255;
        return p;
    };
    const size_t BHLD = (size_t)BH * LL * DD;          // 4,194,304
    float*  qws   = (float*) carve(BHLD * sizeof(float));
    float*  kws   = (float*) carve(BHLD * sizeof(float));
    float*  vws   = (float*) carve(BHLD * sizeof(float));
    float*  vc    = (float*) carve(BHLD * sizeof(float));
    float*  disc  = (float*) carve((size_t)BH * LL * sizeof(float));
    float*  S     = (float*) carve((size_t)BH * TOPP * LL * sizeof(float));
    float*  upd   = (float*) carve((size_t)BH * TOPP * DD * sizeof(float));
    int*    tidx  = (int*)   carve((size_t)BH * TOPP * sizeof(int));
    __bf16* Wt    = (__bf16*)carve((size_t)4 * EE * EE * sizeof(__bf16));
    __bf16* Xbf   = (__bf16*)carve((size_t)3 * BL * EE * sizeof(__bf16));
    __bf16* kbf   = (__bf16*)carve(BHLD * sizeof(__bf16));
    __bf16* vtb   = (__bf16*)carve(BHLD * sizeof(__bf16));
    __bf16* qimpbf= (__bf16*)carve((size_t)BH * TOPP * DD * sizeof(__bf16));
    __bf16* Pbf   = (__bf16*)carve((size_t)BH * TOPP * LL * sizeof(__bf16));
    __bf16* vcbf  = (__bf16*)carve((size_t)BL * EE * sizeof(__bf16));
    (void)ws_size; (void)in_sizes; (void)n_in; (void)out_size;

    // 0) one-time bf16 conversions (weights transposed)
    pa_conv_weights<<<dim3(EE, 4), 256, 0, stream>>>(Wq, Wk, Wv, Wo, Wt);
    pa_conv_x<<<dim3(BL, 3), 256, 0, stream>>>(queries, keys, values, Xbf);
    // 1) Q/K/V projections (WMMA, async-LDS double buffered when available)
    pa_qkv_gemm<<<dim3(EE / 32, BL / 32, 3), 32, 0, stream>>>(
        Xbf, Wt, bq, bk, bv, qws, kws, vws, kbf, vtb);
    // 2) sampled scores
    pa_sample_scores<<<dim3(LL, HH, BB), 32, 0, stream>>>(qws, kws, rand_index, disc);
    // 3) top-35 per (b,h)
    pa_topk<<<dim3(BH), 256, 0, stream>>>(disc, tidx);
    // 4) gather selected queries (bf16, padded)
    pa_gather_q<<<dim3(TOPP, BH), 64, 0, stream>>>(qws, tidx, qimpbf);
    // 5) S = scale * Qimp K^T (WMMA)
    pa_scores_wmma<<<dim3(LL / 64, TOPP / 16, BH), 32, 0, stream>>>(qimpbf, kbf, S);
    // 6) softmax rows (+ bf16 P)
    pa_softmax_rows<<<dim3(BH * TOPP), 256, 0, stream>>>(S, Pbf);
    // 7) upd = P V (WMMA)
    pa_upd_wmma<<<dim3(TOPP / 16, BH), 32, 0, stream>>>(Pbf, vtb, upd);
    // 8) vc = cumsum(v)
    pa_cumsum<<<dim3(BH), 64, 0, stream>>>(vws, vc);
    // 9) scatter updates
    pa_scatter<<<dim3(TOP, BH), 64, 0, stream>>>(tidx, upd, vc);
    // 9b) pack vc rows to bf16
    pa_pack_vc<<<dim3(LL, BB), 256, 0, stream>>>(vc, vcbf);
    // 10) output projection (WMMA)
    pa_out_gemm<<<dim3(EE / 32, BL / 32), 32, 0, stream>>>(vcbf, Wt + (size_t)3 * EE * EE, bo, out);
}
